// UGMPotentialModel_56599079027205
// MI455X (gfx1250) — matrix operand order repaired and checked by reading, not compiled, not run on hardware
//
#include <hip/hip_runtime.h>
#include <hip/hip_bf16.h>

// ---------------------------------------------------------------------------
// UGM potential model on gfx1250 (MI455X).
// Sizes (fixed by the reference): B=4, N=20000, E=320000, C=128, H=64, L=3.
// All dense 64x64 GEMMs run on v_wmma_f32_16x16x32_bf16 (wave32, 16-row tiles).
// ---------------------------------------------------------------------------

typedef __attribute__((ext_vector_type(16))) __bf16 v16bf;
typedef __attribute__((ext_vector_type(8)))  __bf16 v8bf;
typedef __attribute__((ext_vector_type(8)))  float  v8f;

#define BQ   4
#define NN   20000
#define EE   320000
#define CC   128
#define HH   64
#define LL   3
#define ROWS (NN*BQ)      /* 80000 (node-major, batch inner) */
#define TILES (ROWS/16)   /* 5000 exact 16-row WMMA tiles    */

__device__ __forceinline__ float lrelu(float v){ return v > 0.f ? v : 0.01f*v; }

// A-fragment (16x32 bf16) from a row-major [16][64] bf16 LDS tile.
// CDNA5 layout: lane = M (l%16), halves of the wave split K; per lane the 16
// elements are two runs of 8 consecutive K values -> two 16B ds loads.
__device__ __forceinline__ v16bf load_a_frag(const __bf16* tile, int lane, int kt){
  int m = lane & 15, half = lane >> 4;
  const __bf16* p = tile + m*HH + kt*32 + half*8;
  union { v16bf v; v8bf h[2]; } u;
  u.h[0] = *(const v8bf*)(p);        // K = kt*32 + half*8 + [0..7]
  u.h[1] = *(const v8bf*)(p + 16);   // K = kt*32 + 16 + half*8 + [0..7]
  return u.v;
}

__device__ __forceinline__ v8f wmma_bf16(v16bf a, v16bf b, v8f c){
  return __builtin_amdgcn_wmma_f32_16x16x32_bf16(false, a, false, b, (short)0, c, false, false);
}

// ---------------------------------------------------------------------------
__global__ void k_zero(float* p, long n){
  long i = (long)blockIdx.x*blockDim.x + threadIdx.x;
  if (i < n) p[i] = 0.f;
}

// Pre-permute the nine 64x64 f32 weight matrices into per-lane bf16 WMMA
// B-fragments: frag[(mat*2+kt)*4+nt][lane][i] = W[kt*32 + (lane/16)*8 + (i%8) + (i/8)*16][nt*16 + lane%16]
__global__ void k_prep_wfrag(const float* emb_w2, const float* sage_wl, const float* sage_wr,
                             const float* out_w1, const float* out_w2, __bf16* wfrag){
  int tid = blockIdx.x*blockDim.x + threadIdx.x;
  if (tid >= 9*2*4*32*16) return;
  int i    = tid & 15;
  int lane = (tid >> 4) & 31;
  int nt   = (tid >> 9) & 3;
  int kt   = (tid >> 11) & 1;
  int mat  = tid >> 12;
  int n = nt*16 + (lane & 15);
  int k = kt*32 + (lane >> 4)*8 + (i & 7) + (i >> 3)*16;
  const float* W;
  if      (mat == 0) W = emb_w2;
  else if (mat <= 3) W = sage_wl + (size_t)(mat-1)*HH*HH;
  else if (mat <= 6) W = sage_wr + (size_t)(mat-4)*HH*HH;
  else if (mat == 7) W = out_w1;
  else               W = out_w2;
  wfrag[tid] = (__bf16)W[k*HH + n];
}

__global__ void k_deg(const int* ei, float* deg){
  int e = blockIdx.x*blockDim.x + threadIdx.x;
  if (e < EE) atomicAdd(&deg[ei[EE + e]], 1.0f);
}

// h0 = lrelu(lrelu(x*w1+b1) @ w2 + b2), rows are (node,batch) pairs.
__global__ void __launch_bounds__(128) k_embed(const float* x, const float* w1, const float* b1,
                                               const float* b2, const __bf16* wfrag, float* h0){
  __shared__ __bf16 sm[4][16*HH];
  int wv = threadIdx.x >> 5, lane = threadIdx.x & 31;
  int row0 = (blockIdx.x*4 + wv)*16;
  __bf16* tl = sm[wv];
  // Stage the hidden layer (16 rows x 64) in bf16; two lanes per row.
  int r = lane >> 1, c0 = (lane & 1)*32;
  int rho = row0 + r;
  float xv = x[(rho & 3)*NN + (rho >> 2)];
  for (int j = 0; j < 32; ++j){
    int k = c0 + j;
    tl[r*HH + k] = (__bf16)lrelu(xv*w1[k] + b1[k]);
  }
  __syncthreads();
  v16bf a0 = load_a_frag(tl, lane, 0);
  v16bf a1 = load_a_frag(tl, lane, 1);
  const v16bf* wf = (const v16bf*)wfrag;           // mat 0
  int col = lane & 15, half = lane >> 4;
  for (int nt = 0; nt < 4; ++nt){
    float bb = b2[nt*16 + col];
    v8f acc = { bb,bb,bb,bb,bb,bb,bb,bb };
    acc = wmma_bf16(a0, wf[(0*4+nt)*32 + lane], acc);
    acc = wmma_bf16(a1, wf[(1*4+nt)*32 + lane], acc);
    int m0 = row0 + half*8;
    for (int q = 0; q < 8; ++q)
      h0[(size_t)(m0+q)*HH + nt*16 + col] = lrelu(acc[q]);
  }
}

// Edge scatter: agg[dst] += h[src], 1KB per edge, float4 + f32 atomics.
__global__ void k_scatter(const int* ei, const float* h, float* agg){
  long t = (long)blockIdx.x*blockDim.x + threadIdx.x;
  long e = t >> 6;
  int  q = (int)(t & 63);
  if (e >= EE) return;
  int s = ei[e], d = ei[EE + e];
  float4 v = *(const float4*)(h + (size_t)s*256 + q*4);
  float* o = agg + (size_t)d*256 + q*4;
  atomicAdd(o+0, v.x); atomicAdd(o+1, v.y);
  atomicAdd(o+2, v.z); atomicAdd(o+3, v.w);
}

// h' = lrelu((agg/deg) @ Wl + bl + h @ Wr) [+ h0 residual on layer 0]
__global__ void __launch_bounds__(128) k_sage(const float* hin, const float* agg, const float* deg,
                                              const __bf16* wfrag, int matL, int matR,
                                              const float* bias, const float* h0, int add_res,
                                              float* hout){
  __shared__ __bf16 sm[4][2][16*HH];
  int wv = threadIdx.x >> 5, lane = threadIdx.x & 31;
  int row0 = (blockIdx.x*4 + wv)*16;
  __bf16* tA = sm[wv][0];
  __bf16* tB = sm[wv][1];
  int r = lane >> 1, c0 = (lane & 1)*32;
  int rho = row0 + r;
  float dv = deg[rho >> 2];
  float inv = 1.0f / (dv > 1.0f ? dv : 1.0f);
  const float4* pa = (const float4*)(agg + (size_t)rho*HH + c0);
  const float4* ph = (const float4*)(hin + (size_t)rho*HH + c0);
  for (int j = 0; j < 8; ++j){
    float4 va = pa[j], vh = ph[j];
    int base = r*HH + c0 + j*4;
    tA[base+0]=(__bf16)(va.x*inv); tA[base+1]=(__bf16)(va.y*inv);
    tA[base+2]=(__bf16)(va.z*inv); tA[base+3]=(__bf16)(va.w*inv);
    tB[base+0]=(__bf16)vh.x;       tB[base+1]=(__bf16)vh.y;
    tB[base+2]=(__bf16)vh.z;       tB[base+3]=(__bf16)vh.w;
  }
  __syncthreads();
  v16bf aL0 = load_a_frag(tA, lane, 0), aL1 = load_a_frag(tA, lane, 1);
  v16bf aR0 = load_a_frag(tB, lane, 0), aR1 = load_a_frag(tB, lane, 1);
  const v16bf* wf = (const v16bf*)wfrag;
  int col = lane & 15, half = lane >> 4;
  for (int nt = 0; nt < 4; ++nt){
    float bb = bias[nt*16 + col];
    v8f acc = { bb,bb,bb,bb,bb,bb,bb,bb };
    acc = wmma_bf16(aL0, wf[((matL*2+0)*4+nt)*32 + lane], acc);
    acc = wmma_bf16(aL1, wf[((matL*2+1)*4+nt)*32 + lane], acc);
    acc = wmma_bf16(aR0, wf[((matR*2+0)*4+nt)*32 + lane], acc);
    acc = wmma_bf16(aR1, wf[((matR*2+1)*4+nt)*32 + lane], acc);
    int m0 = row0 + half*8;
    for (int q = 0; q < 8; ++q){
      size_t idx = (size_t)(m0+q)*HH + nt*16 + col;
      float v = lrelu(acc[q]);
      if (add_res) v += h0[idx];
      hout[idx] = v;
    }
  }
}

// emb = lrelu(h @ out_w1 + b1) @ out_w2 + b2  (two chained WMMA GEMMs)
__global__ void __launch_bounds__(128) k_outmlp(const float* hin, const __bf16* wfrag,
                                                const float* ob1, const float* ob2, float* emb){
  __shared__ __bf16 sm[4][2][16*HH];
  int wv = threadIdx.x >> 5, lane = threadIdx.x & 31;
  int row0 = (blockIdx.x*4 + wv)*16;
  __bf16* tH = sm[wv][0];
  __bf16* tT = sm[wv][1];
  int r = lane >> 1, c0 = (lane & 1)*32;
  int rho = row0 + r;
  const float4* ph = (const float4*)(hin + (size_t)rho*HH + c0);
  for (int j = 0; j < 8; ++j){
    float4 vh = ph[j];
    int base = r*HH + c0 + j*4;
    tH[base+0]=(__bf16)vh.x; tH[base+1]=(__bf16)vh.y;
    tH[base+2]=(__bf16)vh.z; tH[base+3]=(__bf16)vh.w;
  }
  __syncthreads();
  v16bf a0 = load_a_frag(tH, lane, 0), a1 = load_a_frag(tH, lane, 1);
  const v16bf* wf = (const v16bf*)wfrag;
  int col = lane & 15, half = lane >> 4;
  // layer 1 (mat 7) -> bf16 tile in LDS
  for (int nt = 0; nt < 4; ++nt){
    float bb = ob1[nt*16 + col];
    v8f acc = { bb,bb,bb,bb,bb,bb,bb,bb };
    acc = wmma_bf16(a0, wf[((7*2+0)*4+nt)*32 + lane], acc);
    acc = wmma_bf16(a1, wf[((7*2+1)*4+nt)*32 + lane], acc);
    for (int q = 0; q < 8; ++q)
      tT[(half*8+q)*HH + nt*16 + col] = (__bf16)lrelu(acc[q]);
  }
  __syncthreads();
  v16bf t0 = load_a_frag(tT, lane, 0), t1 = load_a_frag(tT, lane, 1);
  // layer 2 (mat 8), no activation
  for (int nt = 0; nt < 4; ++nt){
    float bb = ob2[nt*16 + col];
    v8f acc = { bb,bb,bb,bb,bb,bb,bb,bb };
    acc = wmma_bf16(t0, wf[((8*2+0)*4+nt)*32 + lane], acc);
    acc = wmma_bf16(t1, wf[((8*2+1)*4+nt)*32 + lane], acc);
    int m0 = row0 + half*8;
    for (int q = 0; q < 8; ++q)
      emb[(size_t)(m0+q)*HH + nt*16 + col] = acc[q];
  }
}

// Per-clique 192->32->16->1 MLP; one wave per (batch, clique).
__global__ void __launch_bounds__(128) k_clique(const float* emb, const int* cliques,
                                                const float* w1, const float* bb1,
                                                const float* w2, const float* bb2,
                                                const float* w3, const float* bb3,
                                                float* out){
  __shared__ float sm[4][192 + 32 + 16];
  int wv = threadIdx.x >> 5, lane = threadIdx.x & 31;
  int idx = blockIdx.x*4 + wv;          // 0..511
  int b = idx >> 7, c = idx & 127;
  float* cin = sm[wv];
  float* s1  = cin + 192;
  float* s2  = s1 + 32;
  for (int t = lane; t < 192; t += 32){
    int node = cliques[c*3 + (t >> 6)];
    cin[t] = emb[((size_t)node*4 + b)*64 + (t & 63)];
  }
  __syncthreads();
  {
    float acc = bb1[c*32 + lane];
    for (int i = 0; i < 192; ++i) acc += cin[i] * w1[((size_t)c*192 + i)*32 + lane];
    s1[lane] = lrelu(acc);
  }
  __syncthreads();
  if (lane < 16){
    float acc = bb2[c*16 + lane];
    for (int i = 0; i < 32; ++i) acc += s1[i] * w2[((size_t)c*32 + i)*16 + lane];
    s2[lane] = lrelu(acc);
  }
  __syncthreads();
  if (lane == 0){
    float acc = bb3[c];
    for (int i = 0; i < 16; ++i) acc += s2[i] * w3[(size_t)c*16 + i];
    out[4 + b*CC + c] = acc;           // per-clique energy
  }
}

__global__ void k_total(float* out){
  __shared__ float red[128];
  int b = blockIdx.x, t = threadIdx.x;
  red[t] = out[4 + b*CC + t];
  __syncthreads();
  for (int s = 64; s > 0; s >>= 1){
    if (t < s) red[t] += red[t + s];
    __syncthreads();
  }
  if (t == 0) out[b] = red[0];
}

// ---------------------------------------------------------------------------
extern "C" void kernel_launch(void* const* d_in, const int* in_sizes, int n_in,
                              void* d_out, int out_size, void* d_ws, size_t ws_size,
                              hipStream_t stream){
  (void)in_sizes; (void)n_in; (void)out_size; (void)ws_size;
  const float* x       = (const float*)d_in[0];
  const int*   ei      = (const int*)d_in[1];
  const int*   cliques = (const int*)d_in[2];
  const float* emb_w1  = (const float*)d_in[3];
  const float* emb_b1  = (const float*)d_in[4];
  const float* emb_w2  = (const float*)d_in[5];
  const float* emb_b2  = (const float*)d_in[6];
  const float* sage_wl = (const float*)d_in[7];
  const float* sage_bl = (const float*)d_in[8];
  const float* sage_wr = (const float*)d_in[9];
  const float* out_w1  = (const float*)d_in[10];
  const float* out_b1  = (const float*)d_in[11];
  const float* out_w2  = (const float*)d_in[12];
  const float* out_b2  = (const float*)d_in[13];
  const float* cl_w1   = (const float*)d_in[14];
  const float* cl_b1   = (const float*)d_in[15];
  const float* cl_w2   = (const float*)d_in[16];
  const float* cl_b2   = (const float*)d_in[17];
  const float* cl_w3   = (const float*)d_in[18];
  const float* cl_b3   = (const float*)d_in[19];
  float* out = (float*)d_out;

  char* ws = (char*)d_ws;
  size_t off = 0;
  auto alloc = [&](size_t bytes)->char*{
    char* p = ws + off;
    off = (off + bytes + 255) & ~(size_t)255;
    return p;
  };
  __bf16* wfrag = (__bf16*)alloc(36864*sizeof(__bf16));
  float*  deg   = (float*) alloc((size_t)NN*sizeof(float));
  size_t hbytes = (size_t)ROWS*HH*sizeof(float);
  float* h0  = (float*)alloc(hbytes);
  float* hA  = (float*)alloc(hbytes);
  float* hB  = (float*)alloc(hbytes);
  float* agg = (float*)alloc(hbytes);
  float* emb = (float*)alloc(hbytes);

  k_prep_wfrag<<<(36864+255)/256, 256, 0, stream>>>(emb_w2, sage_wl, sage_wr, out_w1, out_w2, wfrag);
  k_zero<<<(NN+255)/256, 256, 0, stream>>>(deg, NN);
  k_deg<<<(EE+255)/256, 256, 0, stream>>>(ei, deg);
  k_embed<<<TILES/4, 128, 0, stream>>>(x, emb_w1, emb_b1, emb_b2, wfrag, h0);

  const float* hin = h0;
  float* hout = hA;
  for (int l = 0; l < LL; ++l){
    k_zero<<<((long)ROWS*HH + 255)/256, 256, 0, stream>>>(agg, (long)ROWS*HH);
    k_scatter<<<(EE*64)/256, 256, 0, stream>>>(ei, hin, agg);
    k_sage<<<TILES/4, 128, 0, stream>>>(hin, agg, deg, wfrag, 1+l, 4+l,
                                        sage_bl + (size_t)l*HH, h0, (l==0)?1:0, hout);
    hin = hout;
    hout = (hout == hA) ? hB : hA;
  }
  k_outmlp<<<TILES/4, 128, 0, stream>>>(hin, wfrag, out_b1, out_b2, emb);
  k_clique<<<(BQ*CC)/4, 128, 0, stream>>>(emb, cliques, cl_w1, cl_b1, cl_w2, cl_b2, cl_w3, cl_b3, out);
  k_total<<<BQ, 128, 0, stream>>>(out);
}
